// PLE_1297080123437
// MI455X (gfx1250) — matrix operand order repaired and checked by reading, not compiled, not run on hardware
//
#include <hip/hip_runtime.h>
#include <hip/hip_bf16.h>
#include <math.h>

// ---------------------------------------------------------------------------
// Problem constants (from reference)
// ---------------------------------------------------------------------------
constexpr int kB    = 16384;
constexpr int kF    = 18;
constexpr int kV    = 100000;
constexpr int kE    = 64;
constexpr int kIN0  = kF * kE;      // 1152
constexpr int kD1   = 512;
constexpr int kD2   = 256;
constexpr int kNEXP = 12;

// ---------------------------------------------------------------------------
// bf16 helpers (bit-level; __bf16 used only for WMMA operand vectors)
// ---------------------------------------------------------------------------
__device__ __forceinline__ unsigned short f2bf(float x) {
    unsigned int u = __float_as_uint(x);
    unsigned int r = (u + 0x7FFFu + ((u >> 16) & 1u)) >> 16;   // RNE
    return (unsigned short)r;
}
__device__ __forceinline__ float bf2f(unsigned short h) {
    return __uint_as_float(((unsigned int)h) << 16);
}

typedef __attribute__((ext_vector_type(16))) __bf16        v16bf;
typedef __attribute__((ext_vector_type(8)))  __bf16        v8bf;
typedef __attribute__((ext_vector_type(8)))  float         v8f;
typedef __attribute__((ext_vector_type(4)))  unsigned int  v4u;
typedef __attribute__((ext_vector_type(8)))  int           v8i;
typedef __attribute__((ext_vector_type(4)))  int           v4i;

#if __has_builtin(__builtin_amdgcn_tensor_load_to_lds) && \
    __has_builtin(__builtin_amdgcn_s_wait_tensorcnt)
#define USE_TDM 1
#else
#define USE_TDM 0
#endif

// ---------------------------------------------------------------------------
// Embedding gather: feat[b, f*64+e] = emb[f, x[b,f], e]  (bf16 out)
// ---------------------------------------------------------------------------
__global__ __launch_bounds__(256) void embed_kernel(
    const int* __restrict__ x, const float* __restrict__ emb,
    unsigned short* __restrict__ feat)
{
    long long idx = (long long)blockIdx.x * 256 + threadIdx.x;
    if (idx >= (long long)kB * kIN0) return;
    int b = (int)(idx / kIN0);
    int r = (int)(idx - (long long)b * kIN0);
    int f = r >> 6;
    int e = r & 63;
    int row = x[b * kF + f];
    feat[idx] = f2bf(emb[((size_t)f * kV + row) * kE + e]);
}

// ---------------------------------------------------------------------------
// Weight convert+transpose: W[s][K][N] f32  ->  Wt[s][N][K] bf16
// ---------------------------------------------------------------------------
__global__ __launch_bounds__(256) void wconv_kernel(
    const float* __restrict__ W, unsigned short* __restrict__ Wt,
    int K, int N, long long total)
{
    long long idx = (long long)blockIdx.x * 256 + threadIdx.x;
    if (idx >= total) return;
    long long KN = (long long)K * N;
    int s = (int)(idx / KN);
    long long rem = idx - (long long)s * KN;
    int n = (int)(rem / K);
    int k = (int)(rem - (long long)n * K);
    Wt[idx] = f2bf(W[(size_t)s * KN + (size_t)k * N + n]);
}

// ---------------------------------------------------------------------------
// WMMA GEMM:  C[M,N](bf16) = A[M,K](bf16,row-major) * Wt[N,K](bf16,row-major)
//
// Block = 8 waves, block tile 128(M) x 64(N); wave tile 16x64 ->
// 4x v_wmma_f32_16x16x32_bf16 per 32-wide K-step.
// The 64xK weight panel (shared by all 8 waves) is staged into LDS once per
// block with the Tensor Data Mover (tensor_load_to_lds + s_wait_tensorcnt),
// then B fragments come from ds_load_b128 while the A stream uses global
// b128 loads + prefetch.
//
// Fragment layouts per CDNA5 ISA 7.12.2:
//   A 16x32: lane&15 = row; half-wave selects K {0..7,16..23} vs {8..15,24..31}
//   B 32x16: lane&15 = col; half-wave selects K {0..7,16..23} vs {8..15,24..31}
// Requires M%128==0, N%64==0, K%32==0.
// ---------------------------------------------------------------------------
__global__ __launch_bounds__(256) void gemm_bf16_kernel(
    const unsigned short* __restrict__ A,
    const unsigned short* __restrict__ Wt,
    unsigned short* __restrict__ C,
    int M, int N, int K)
{
    extern __shared__ unsigned short smemB[];   // [64][K] bf16 weight panel

    const int lane = threadIdx.x & 31;
    const int wave = threadIdx.x >> 5;
    const int n0   = blockIdx.x * 64;
    const int m0   = blockIdx.y * 128 + wave * 16;
    const int half = lane >> 4;          // 0 or 1
    const int l16  = lane & 15;

    // ---- stage the 64xK weight panel into LDS ----
#if USE_TDM
    if (threadIdx.x < 32) {              // wave 0 issues one TDM descriptor
        const unsigned long long ga =
            (unsigned long long)(uintptr_t)(Wt + (size_t)n0 * K);
        // LDS aperture: low 32 bits of a generic shared pointer = LDS byte off
        const unsigned int ldsOff = (unsigned int)(uintptr_t)(void*)smemB;
        // D# group 0: count=1 | lds_addr | global_addr[56:0] | type=2
        v4u g0 = { 1u,
                   ldsOff,
                   (unsigned int)(ga & 0xFFFFFFFFu),
                   (unsigned int)((ga >> 32) & 0x01FFFFFFu) | 0x80000000u };
        // D# group 1: data_size=2B; tensor = K x 64 (stride K);
        // tile = K x 64  (dims in data_size units)
        v8i g1 = { (int)0x00010000u,                    // mask=0, data_size=1(2B)
                   (int)(((unsigned)K & 0xFFFFu) << 16),// tensor_dim0 lo16 @bit48
                   (int)(64u << 16),                    // dim0 hi=0, tensor_dim1=64
                   (int)(((unsigned)K & 0xFFFFu) << 16),// dim1 hi=0, tile_dim0=K
                   64,                                  // tile_dim1=64, tile_dim2=0
                   K,                                   // tensor_dim0_stride lo32
                   0, 0 };                              // stride hi, dim1_stride=0
        v4i gz4  = { 0, 0, 0, 0 };
        v8i gz8  = { 0, 0, 0, 0, 0, 0, 0, 0 };
        (void)gz8;
#if defined(__clang_major__) && (__clang_major__ >= 23)
        __builtin_amdgcn_tensor_load_to_lds(g0, g1, gz4, gz4, gz8, 0);
#else
        __builtin_amdgcn_tensor_load_to_lds(g0, g1, gz4, gz4, 0);
#endif
        __builtin_amdgcn_s_wait_tensorcnt(0);
    }
#else
    {   // cooperative fallback copy (16B per thread per step)
        const unsigned short* src = Wt + (size_t)n0 * K;
        const int total = 64 * K;
        for (int i = threadIdx.x * 8; i < total; i += 256 * 8)
            *(v8bf*)(smemB + i) = *(const v8bf*)(src + i);
    }
#endif
    __syncthreads();

    const int akb = half * 8;            // A k-base within 32-wide step
    const int bkb = half * 8;            // B k-base within 32-wide step
    const unsigned short* aptr = A + (size_t)(m0 + l16) * K;

    v8f acc[4] = {v8f{}, v8f{}, v8f{}, v8f{}};

    for (int kk = 0; kk < K; kk += 32) {
        if (kk + 32 < K)
            __builtin_prefetch(aptr + kk + 32 + akb, 0, 1);  // global_prefetch_b8
        union { v16bf v; v8bf h[2]; } af;
        af.h[0] = *(const v8bf*)(aptr + kk + akb);        // K = akb..akb+7
        af.h[1] = *(const v8bf*)(aptr + kk + akb + 16);   // K = 16+akb..
#pragma unroll
        for (int j = 0; j < 4; ++j) {
            union { v16bf v; v8bf h[2]; } bfm;
            const unsigned short* bb =
                smemB + (size_t)(j * 16 + l16) * K + kk + bkb;
            bfm.h[0] = *(const v8bf*)(bb);                 // ds_load_b128
            bfm.h[1] = *(const v8bf*)(bb + 16);
            acc[j] = __builtin_amdgcn_wmma_f32_16x16x32_bf16(
                false, af.v, false, bfm.v, (short)0, acc[j], false, false);
        }
    }

#pragma unroll
    for (int j = 0; j < 4; ++j) {
#pragma unroll
        for (int r = 0; r < 8; ++r) {
            int m = m0 + half * 8 + r;                  // C/D layout (ISA table)
            C[(size_t)m * N + n0 + j * 16 + l16] = f2bf(acc[j][r]);
        }
    }
}

// ---------------------------------------------------------------------------
// BN batch-stat accumulation (chunked, atomics) + finalize
// sums layout: [plane 0: sum][plane 1: sumsq], each nslot*C floats.
// ---------------------------------------------------------------------------
__global__ __launch_bounds__(64) void stats_part_kernel(
    const unsigned short* __restrict__ h, float* __restrict__ sums,
    int C, int Brows, int chunkRows)
{
    int col  = blockIdx.x * 64 + threadIdx.x;
    int slot = blockIdx.z;
    int r0   = blockIdx.y * chunkRows;
    const unsigned short* hp = h + ((size_t)slot * Brows + r0) * C + col;
    float s = 0.f, q = 0.f;
    for (int r = 0; r < chunkRows; ++r) {
        float v = bf2f(hp[(size_t)r * C]);
        s += v; q += v * v;
    }
    int planeN = gridDim.z * C;
    atomicAdd(&sums[slot * C + col], s);
    atomicAdd(&sums[planeN + slot * C + col], q);
}

__global__ __launch_bounds__(256) void finalize_stats_kernel(
    float* __restrict__ stats, int planeN, float invRows)
{
    int i = blockIdx.x * 256 + threadIdx.x;
    if (i >= planeN) return;
    float m   = stats[i] * invRows;
    float var = stats[planeN + i] * invRows - m * m;
    stats[i]          = m;
    stats[planeN + i] = rsqrtf(fmaxf(var, 0.f) + 1e-5f);
}

__global__ __launch_bounds__(256) void fill0_kernel(float* p, int n)
{
    int i = blockIdx.x * 256 + threadIdx.x;
    if (i < n) p[i] = 0.f;
}

// ---------------------------------------------------------------------------
// Gate: probs[b, 0..NG) = softmax(X[b,:] @ gw + gb), padded to width 12
// ---------------------------------------------------------------------------
__global__ __launch_bounds__(256) void gate_kernel(
    const unsigned short* __restrict__ X, const float* __restrict__ gw,
    const float* __restrict__ gb, float* __restrict__ probs,
    int K, int NG, int Brows)
{
    int b = blockIdx.x * 256 + threadIdx.x;
    if (b >= Brows) return;
    float lg[12];
#pragma unroll
    for (int e = 0; e < 12; ++e) lg[e] = (e < NG) ? gb[e] : 0.f;
    const unsigned short* xr = X + (size_t)b * K;
    for (int k = 0; k < K; ++k) {
        float xv = bf2f(xr[k]);
        const float* wr = gw + (size_t)k * NG;
#pragma unroll
        for (int e = 0; e < 12; ++e)
            if (e < NG) lg[e] += xv * wr[e];
    }
    float mx = lg[0];
#pragma unroll
    for (int e = 1; e < 12; ++e) if (e < NG) mx = fmaxf(mx, lg[e]);
    float ss = 0.f;
#pragma unroll
    for (int e = 0; e < 12; ++e)
        if (e < NG) { lg[e] = __expf(lg[e] - mx); ss += lg[e]; }
    float inv = 1.f / ss;
    float* pr = probs + (size_t)b * 12;
#pragma unroll
    for (int e = 0; e < 12; ++e) pr[e] = (e < NG) ? lg[e] * inv : 0.f;
}

// ---------------------------------------------------------------------------
// Combine: out[b,o] = sum_i probs[b,i] * relu(BN(h[slot_i][b,o]))
// ---------------------------------------------------------------------------
struct IMap { int v[12]; };

__global__ __launch_bounds__(256) void combine_kernel(
    const unsigned short* __restrict__ h, const float* __restrict__ stats,
    int statPlane, const float* __restrict__ gamma,
    const float* __restrict__ beta, const float* __restrict__ probs,
    unsigned short* __restrict__ outp,
    IMap smap, IMap emap, int nE, int C, int Brows)
{
    int b = blockIdx.x;
    int o = blockIdx.y * 256 + threadIdx.x;
    if (o >= C) return;
    const float* pr = probs + (size_t)b * 12;
    float acc = 0.f;
    for (int i = 0; i < nE; ++i) {
        int s = smap.v[i];
        int e = emap.v[i];
        float hv = bf2f(h[((size_t)s * Brows + b) * C + o]);
        float bn = (hv - stats[s * C + o]) * stats[statPlane + s * C + o] *
                   gamma[(size_t)e * C + o] + beta[(size_t)e * C + o];
        acc += pr[i] * fmaxf(bn, 0.f);
    }
    outp[(size_t)b * C + o] = f2bf(acc);
}

// ---------------------------------------------------------------------------
// Tower BN+ReLU (batched over nslot planes)
// ---------------------------------------------------------------------------
__global__ __launch_bounds__(256) void bnrelu_kernel(
    const unsigned short* __restrict__ z, const float* __restrict__ stats,
    int statPlane, const float* __restrict__ gamma,
    const float* __restrict__ beta, unsigned short* __restrict__ y,
    int C, int BC, long long total)
{
    long long idx = (long long)blockIdx.x * 256 + threadIdx.x;
    if (idx >= total) return;
    int slot = (int)(idx / BC);
    int o    = (int)(idx % C);
    float v = bf2f(z[idx]);
    v = (v - stats[slot * C + o]) * stats[statPlane + slot * C + o] *
        gamma[slot * C + o] + beta[slot * C + o];
    y[idx] = f2bf(fmaxf(v, 0.f));
}

// ---------------------------------------------------------------------------
// Final head: out[t*B+b] = sigmoid( y1[t][b,:] . tow[t] + tob[t] )
// ---------------------------------------------------------------------------
__global__ __launch_bounds__(256) void final_kernel(
    const unsigned short* __restrict__ y1, const float* __restrict__ tow,
    const float* __restrict__ tob, float* __restrict__ outp, int Brows)
{
    int idx = blockIdx.x * 256 + threadIdx.x;
    if (idx >= 2 * Brows) return;
    int t = idx / Brows;
    int b = idx % Brows;
    const unsigned short* yr = y1 + ((size_t)t * Brows + b) * 64;
    const float* w = tow + t * 64;
    float s = tob[t];
#pragma unroll
    for (int o = 0; o < 64; ++o) s += bf2f(yr[o]) * w[o];
    outp[idx] = 1.f / (1.f + __expf(-s));
}

// ---------------------------------------------------------------------------
// Host side
// ---------------------------------------------------------------------------
static inline size_t alignup(size_t x) { return (x + 255) & ~(size_t)255; }

static void launch_gemm(const unsigned short* A, const unsigned short* Wt,
                        unsigned short* C, int M, int N, int K,
                        hipStream_t s)
{
    dim3 g(N / 64, M / 128);
    size_t shmem = (size_t)64 * K * 2;       // B panel in LDS
    gemm_bf16_kernel<<<g, dim3(256), shmem, s>>>(A, Wt, C, M, N, K);
}

extern "C" void kernel_launch(void* const* d_in, const int* in_sizes, int n_in,
                              void* d_out, int out_size, void* d_ws, size_t ws_size,
                              hipStream_t stream)
{
    (void)in_sizes; (void)n_in; (void)out_size; (void)ws_size;

    // ---- inputs (setup_inputs dict order) ----
    const int*   x    = (const int*)  d_in[0];
    const float* emb  = (const float*)d_in[1];
    const float* ew0  = (const float*)d_in[2];
    // eb0 (d_in[3]) absorbed by training-mode BN
    const float* eg0  = (const float*)d_in[4];
    const float* ebt0 = (const float*)d_in[5];
    const float* gw0  = (const float*)d_in[6];
    const float* gb0  = (const float*)d_in[7];
    const float* gsw0 = (const float*)d_in[8];
    const float* gsb0 = (const float*)d_in[9];
    const float* ew1  = (const float*)d_in[10];
    // eb1 absorbed
    const float* eg1  = (const float*)d_in[12];
    const float* ebt1 = (const float*)d_in[13];
    const float* gw1  = (const float*)d_in[14];
    const float* gb1  = (const float*)d_in[15];
    // gsw1/gsb1 (16,17) unused: shared-stream output is dead after layer 1
    const float* tw0  = (const float*)d_in[18];
    // tb0 absorbed
    const float* tg0  = (const float*)d_in[20];
    const float* tbt0 = (const float*)d_in[21];
    const float* tw1  = (const float*)d_in[22];
    // tb1 absorbed
    const float* tg1  = (const float*)d_in[24];
    const float* tbt1 = (const float*)d_in[25];
    const float* tow  = (const float*)d_in[26];
    const float* tob  = (const float*)d_in[27];
    float* outp = (float*)d_out;

    // ---- workspace layout ----
    char* ws = (char*)d_ws;
    size_t off = 0;
    auto alloc = [&](size_t bytes) { size_t o = off; off = alignup(off + bytes); return o; };

    unsigned short* feat  = (unsigned short*)(ws + alloc((size_t)kB * kIN0 * 2));
    unsigned short* ew0t  = (unsigned short*)(ws + alloc((size_t)kNEXP * kD1 * kIN0 * 2));
    unsigned short* ew1t  = (unsigned short*)(ws + alloc((size_t)kNEXP * kD2 * kD1 * 2));
    unsigned short* tw0t  = (unsigned short*)(ws + alloc((size_t)2 * 128 * 256 * 2));
    unsigned short* tw1t  = (unsigned short*)(ws + alloc((size_t)2 * 64 * 128 * 2));
    float* stats0  = (float*)(ws + alloc((size_t)kNEXP * kD1 * 2 * 4));
    float* stats1  = (float*)(ws + alloc((size_t)16 * kD2 * 2 * 4));
    float* statsT0 = (float*)(ws + alloc((size_t)2 * 128 * 2 * 4));
    float* statsT1 = (float*)(ws + alloc((size_t)2 * 64 * 2 * 4));
    float* probs0  = (float*)(ws + alloc((size_t)3 * kB * 12 * 4));
    float* probs1  = (float*)(ws + alloc((size_t)2 * kB * 12 * 4));
    unsigned short* fea0 = (unsigned short*)(ws + alloc((size_t)3 * kB * kD1 * 2));
    unsigned short* fea1 = (unsigned short*)(ws + alloc((size_t)2 * kB * kD2 * 2));
    unsigned short* z0   = (unsigned short*)(ws + alloc((size_t)2 * kB * 128 * 2));
    unsigned short* y0   = (unsigned short*)(ws + alloc((size_t)2 * kB * 128 * 2));
    unsigned short* z1   = (unsigned short*)(ws + alloc((size_t)2 * kB * 64 * 2));
    unsigned short* y1   = (unsigned short*)(ws + alloc((size_t)2 * kB * 64 * 2));
    unsigned short* h0   = (unsigned short*)(ws + alloc((size_t)kNEXP * kB * kD1 * 2)); // 201MB
    unsigned short* h1   = h0;  // h0 dead before layer-1 GEMMs start; alias (134MB)

    const int TIDX[3][12] = {
        {0,1,2,3,8,9,10,11, 0,0,0,0},
        {4,5,6,7,8,9,10,11, 0,0,0,0},
        {0,1,2,3,4,5,6,7,8,9,10,11}};

    // ---- 1. embedding gather ----
    {
        long long total = (long long)kB * kIN0;
        embed_kernel<<<dim3((unsigned)((total + 255) / 256)), 256, 0, stream>>>(x, emb, feat);
    }

    // ---- 2. weight transpose/convert ----
    {
        long long t0 = (long long)kNEXP * kIN0 * kD1;
        wconv_kernel<<<dim3((unsigned)((t0 + 255) / 256)), 256, 0, stream>>>(ew0, ew0t, kIN0, kD1, t0);
        long long t1 = (long long)kNEXP * kD1 * kD2;
        wconv_kernel<<<dim3((unsigned)((t1 + 255) / 256)), 256, 0, stream>>>(ew1, ew1t, kD1, kD2, t1);
        long long t2 = (long long)2 * 256 * 128;
        wconv_kernel<<<dim3((unsigned)((t2 + 255) / 256)), 256, 0, stream>>>(tw0, tw0t, 256, 128, t2);
        long long t3 = (long long)2 * 128 * 64;
        wconv_kernel<<<dim3((unsigned)((t3 + 255) / 256)), 256, 0, stream>>>(tw1, tw1t, 128, 64, t3);
    }

    // ---- 3. layer 0: 12 unique expert GEMMs (streams share inputs) ----
    for (int e = 0; e < kNEXP; ++e)
        launch_gemm(feat, ew0t + (size_t)e * kD1 * kIN0,
                    h0 + (size_t)e * kB * kD1, kB, kD1, kIN0, stream);

    // ---- 4. layer-0 BN stats ----
    {
        int n = kNEXP * kD1 * 2;
        fill0_kernel<<<(n + 255) / 256, 256, 0, stream>>>(stats0, n);
        stats_part_kernel<<<dim3(kD1 / 64, 128, kNEXP), 64, 0, stream>>>(h0, stats0, kD1, kB, kB / 128);
        finalize_stats_kernel<<<(kNEXP * kD1 + 255) / 256, 256, 0, stream>>>(stats0, kNEXP * kD1, 1.f / kB);
    }

    // ---- 5. layer-0 gates ----
    for (int j = 0; j < 2; ++j)
        gate_kernel<<<(kB + 255) / 256, 256, 0, stream>>>(
            feat, gw0 + (size_t)j * kIN0 * 8, gb0 + j * 8,
            probs0 + (size_t)j * kB * 12, kIN0, 8, kB);
    gate_kernel<<<(kB + 255) / 256, 256, 0, stream>>>(
        feat, gsw0, gsb0, probs0 + (size_t)2 * kB * 12, kIN0, 12, kB);

    // ---- 6. layer-0 combine (slot == expert) ----
    for (int j = 0; j < 3; ++j) {
        IMap sm{}, em{};
        int nE = (j < 2) ? 8 : 12;
        for (int i = 0; i < 12; ++i) { sm.v[i] = TIDX[j][i]; em.v[i] = TIDX[j][i]; }
        combine_kernel<<<dim3(kB, kD1 / 256), 256, 0, stream>>>(
            h0, stats0, kNEXP * kD1, eg0, ebt0,
            probs0 + (size_t)j * kB * 12,
            fea0 + (size_t)j * kB * kD1, sm, em, nE, kD1, kB);
    }

    // ---- 7. layer 1: 16 GEMMs (shared-stream output is dead -> skipped) ----
    for (int s = 0; s < 16; ++s) {
        int js = s / 8, loc = s % 8;
        int expert = TIDX[js][loc];
        launch_gemm(fea0 + (size_t)js * kB * kD1,
                    ew1t + (size_t)expert * kD2 * kD1,
                    h1 + (size_t)s * kB * kD2, kB, kD2, kD1, stream);
    }

    // ---- 8. layer-1 BN stats ----
    {
        int n = 16 * kD2 * 2;
        fill0_kernel<<<(n + 255) / 256, 256, 0, stream>>>(stats1, n);
        stats_part_kernel<<<dim3(kD2 / 64, 128, 16), 64, 0, stream>>>(h1, stats1, kD2, kB, kB / 128);
        finalize_stats_kernel<<<(16 * kD2 + 255) / 256, 256, 0, stream>>>(stats1, 16 * kD2, 1.f / kB);
    }

    // ---- 9. layer-1 gates + combine ----
    for (int j = 0; j < 2; ++j)
        gate_kernel<<<(kB + 255) / 256, 256, 0, stream>>>(
            fea0 + (size_t)j * kB * kD1, gw1 + (size_t)j * kD1 * 8, gb1 + j * 8,
            probs1 + (size_t)j * kB * 12, kD1, 8, kB);
    for (int j = 0; j < 2; ++j) {
        IMap sm{}, em{};
        for (int i = 0; i < 12; ++i) { sm.v[i] = j * 8 + (i % 8); em.v[i] = TIDX[j][i % 8]; }
        combine_kernel<<<dim3(kB, 1), 256, 0, stream>>>(
            h1, stats1, 16 * kD2, eg1, ebt1,
            probs1 + (size_t)j * kB * 12,
            fea1 + (size_t)j * kB * kD2, sm, em, 8, kD2, kB);
    }

    // ---- 10. towers ----
    for (int t = 0; t < 2; ++t)
        launch_gemm(fea1 + (size_t)t * kB * kD2, tw0t + (size_t)t * 128 * 256,
                    z0 + (size_t)t * kB * 128, kB, 128, 256, stream);
    {
        int n = 2 * 128 * 2;
        fill0_kernel<<<(n + 255) / 256, 256, 0, stream>>>(statsT0, n);
        stats_part_kernel<<<dim3(128 / 64, 128, 2), 64, 0, stream>>>(z0, statsT0, 128, kB, kB / 128);
        finalize_stats_kernel<<<(2 * 128 + 255) / 256, 256, 0, stream>>>(statsT0, 2 * 128, 1.f / kB);
        long long total = (long long)2 * kB * 128;
        bnrelu_kernel<<<(unsigned)((total + 255) / 256), 256, 0, stream>>>(
            z0, statsT0, 2 * 128, tg0, tbt0, y0, 128, kB * 128, total);
    }
    for (int t = 0; t < 2; ++t)
        launch_gemm(y0 + (size_t)t * kB * 128, tw1t + (size_t)t * 64 * 128,
                    z1 + (size_t)t * kB * 64, kB, 64, 128, stream);
    {
        int n = 2 * 64 * 2;
        fill0_kernel<<<(n + 255) / 256, 256, 0, stream>>>(statsT1, n);
        stats_part_kernel<<<dim3(1, 128, 2), 64, 0, stream>>>(z1, statsT1, 64, kB, kB / 128);
        finalize_stats_kernel<<<(2 * 64 + 255) / 256, 256, 0, stream>>>(statsT1, 2 * 64, 1.f / kB);
        long long total = (long long)2 * kB * 64;
        bnrelu_kernel<<<(unsigned)((total + 255) / 256), 256, 0, stream>>>(
            z1, statsT1, 2 * 64, tg1, tbt1, y1, 64, kB * 64, total);
    }

    // ---- 11. heads + sigmoid ----
    final_kernel<<<(2 * kB + 255) / 256, 256, 0, stream>>>(y1, tow, tob, outp, kB);
}
